// FarMOS_41283225649436
// MI455X (gfx1250) — matrix operand order repaired and checked by reading, not compiled, not run on hardware
//
#include <hip/hip_runtime.h>
#include <hip/hip_bf16.h>

typedef __attribute__((ext_vector_type(16))) _Float16 v16h;
typedef __attribute__((ext_vector_type(8)))  _Float16 v8h;
typedef __attribute__((ext_vector_type(8)))  float    v8f;
typedef __attribute__((ext_vector_type(4)))  unsigned int u32x4;
typedef __attribute__((ext_vector_type(8)))  int      i32x8;
typedef __attribute__((ext_vector_type(4)))  int      i32x4;

// Problem constants (match reference)
#define KB   2          // batch
#define KT   2          // time
#define KN   131072     // points
#define KBT  (KB*KT)

#define TDM_OK __has_builtin(__builtin_amdgcn_tensor_load_to_lds)

// ---------------------------------------------------------------------------
// CDNA5 WMMA fragment helpers (16x16x32 f16 -> f32), wave32; A/B use this file's contiguous k order (not ISA table order).
// A (16x32 f16): M = lane&15; K = (lane&16?16:0) + e
//   -> one contiguous 16-half run per lane when weights are row-major [M][K].
// B (32x16 f16): N = lane&15; K = (lane&16?16:0) + e
//   -> one contiguous 16-half run per lane when activations are [N][K].
// D (16x16 f32): M = r + 8*(lane>=16); N = lane&15
// ---------------------------------------------------------------------------
__device__ __forceinline__ v16h load_a_frag(const _Float16* Wm, int m0, int k0,
                                            int ldk, int lane) {
  v16h a;
  const int mrow = m0 + (lane & 15);
  const int kb = (lane & 16) ? 16 : 0;
#pragma unroll
  for (int e = 0; e < 16; ++e) {
    const int k = k0 + kb + e;
    a[e] = Wm[mrow * ldk + k];
  }
  return a;
}

// Activations stored [col][K] (K contiguous): one 32B contiguous read.
__device__ __forceinline__ v16h load_b_frag_kcontig(const _Float16* bp) {
  v16h b;
#pragma unroll
  for (int e = 0; e < 16; ++e) b[e] = bp[e];
  return b;
}

// ---------------------------------------------------------------------------
// PointNet: f = relu(W2 * relu(W1*x + b1) + b2) * valid, via f16 WMMA.
// Block: 256 threads = 8 waves -> 64 out-ch x 32 points. Grid: (N/32, B*T).
// Activations kept [point][K] in LDS so B fragments are ds_load_b128 pairs.
// ---------------------------------------------------------------------------
__global__ void pointnet_wmma_kernel(const float* __restrict__ xyzi,
                                     const float* __restrict__ w1,
                                     const float* __restrict__ b1,
                                     const float* __restrict__ w2,
                                     const float* __restrict__ b2,
                                     float* __restrict__ f) {
  __shared__ __align__(16) _Float16 sW1[64 * 32];   // [m][k], K padded 7->32
  __shared__ __align__(16) _Float16 sW2[64 * 64];   // [m][k]
  __shared__ __align__(16) _Float16 sX[32 * 32];    // [pt][k]
  __shared__ __align__(16) _Float16 sH[32 * 64];    // [pt][k]
  __shared__ float    sValid[32];
  __shared__ float    sB1[64];
  __shared__ float    sB2[64];

  const int tid  = threadIdx.x;
  const int lane = tid & 31;
  const int wave = tid >> 5;
  const int bt = blockIdx.y;
  const int n0 = blockIdx.x * 32;

  for (int i = tid; i < 64 * 32; i += blockDim.x) {
    const int m = i >> 5, k = i & 31;
    sW1[i] = (k < 7) ? (_Float16)w1[m * 7 + k] : (_Float16)0.f;
  }
  for (int i = tid; i < 64 * 64; i += blockDim.x)
    sW2[i] = (_Float16)w2[i];
  if (tid < 64) { sB1[tid] = b1[tid]; sB2[tid] = b2[tid]; }
  if (tid < 32) {
    const float range = xyzi[((long long)bt * 7 + 4) * KN + n0 + tid];
    sValid[tid] = (range < 100.0f) ? 1.0f : 0.0f;
  }
  __syncthreads();

  for (int i = tid; i < 32 * 32; i += blockDim.x) {
    const int p = i >> 5, k = i & 31;
    float v = 0.0f;
    if (k < 7) v = xyzi[((long long)bt * 7 + k) * KN + n0 + p] * sValid[p];
    sX[p * 32 + k] = (_Float16)v;
  }
  __syncthreads();

  const int ch_tile = wave & 3;   // 4 tiles of 16 out-channels
  const int pt_tile = wave >> 2;  // 2 tiles of 16 points
  const int kbB = (lane & 16) ? 16 : 0;
  const int col = pt_tile * 16 + (lane & 15);
  const int hi8 = (lane & 16) ? 8 : 0;

  // ---- Layer 1: one WMMA per tile (K padded to 32) ----
  {
    v8f acc = {};
    v16h a = load_a_frag(sW1, ch_tile * 16, 0, 32, lane);
    v16h b = load_b_frag_kcontig(&sX[col * 32 + kbB]);
    acc = __builtin_amdgcn_wmma_f32_16x16x32_f16(false, a, false, b,
                                                 (short)0, acc, false, false);
    v8h hv;
#pragma unroll
    for (int r = 0; r < 8; ++r) {
      const int m = ch_tile * 16 + hi8 + r;
      hv[r] = (_Float16)fmaxf(acc[r] + sB1[m], 0.0f);
    }
    // 8 consecutive K slots of this point's hidden vector: one 16B store.
    *(v8h*)&sH[col * 64 + ch_tile * 16 + hi8] = hv;
  }
  __syncthreads();

  // ---- Layer 2: two WMMAs per tile (K = 64) ----
  {
    v8f acc = {};
#pragma unroll
    for (int k0 = 0; k0 < 64; k0 += 32) {
      v16h a = load_a_frag(sW2, ch_tile * 16, k0, 64, lane);
      v16h b = load_b_frag_kcontig(&sH[col * 64 + k0 + kbB]);
      acc = __builtin_amdgcn_wmma_f32_16x16x32_f16(false, a, false, b,
                                                   (short)0, acc, false, false);
    }
#pragma unroll
    for (int r = 0; r < 8; ++r) {
      const int m = ch_tile * 16 + hi8 + r;
      const float v = fmaxf(acc[r] + sB2[m], 0.0f) * sValid[col];
      f[((long long)bt * 64 + m) * KN + n0 + col] = v;
    }
  }
}

// ---------------------------------------------------------------------------
// Scatter-max into BEV over all (b,t) points (post-ReLU features >= 0 so
// float-max == int-max on bit patterns; grids zero-initialized, matching the
// reference's isfinite->0 handling).
// ---------------------------------------------------------------------------
__global__ void scatter_bev_all_kernel(const float* __restrict__ f,
                                       const float* __restrict__ des,
                                       float* __restrict__ bev) {
  const long long idx = (long long)blockIdx.x * blockDim.x + threadIdx.x;
  if (idx >= (long long)KBT * KN) return;
  const int bt = (int)(idx / KN);
  const int n  = (int)(idx % KN);
  const int b  = bt / KT;
  const float* cp = des + ((long long)bt * KN + n) * 3;
  const int ri = min(max((int)floorf(cp[0]), 0), 511);
  const int ci = min(max((int)floorf(cp[1]), 0), 511);
  const long long cell = (long long)ri * 512 + ci;
#pragma unroll 4
  for (int ch = 0; ch < 64; ++ch) {
    const float v = f[((long long)bt * 64 + ch) * KN + n];
    atomicMax((int*)&bev[((long long)(b * 64 + ch)) * (512 * 512) + cell],
              __float_as_int(v));
  }
}

// Generic scatter-max for t = T-1 features: feat (B, C, N), batch stride
// fbstride; coords indexed (b, T-1, n, comp).
__global__ void scatter_t0_kernel(const float* __restrict__ feat,
                                  long long fbstride, int C,
                                  const float* __restrict__ coord,
                                  long long cbstride, long long ctoff,
                                  int comp_r, int comp_c, float scale,
                                  int H, int W, float* __restrict__ out) {
  const long long idx = (long long)blockIdx.x * blockDim.x + threadIdx.x;
  if (idx >= (long long)KB * KN) return;
  const int b = (int)(idx / KN);
  const int n = (int)(idx % KN);
  const float* cp = coord + (long long)b * cbstride + ctoff + (long long)n * 3;
  const int ri = min(max((int)floorf(cp[comp_r] * scale), 0), H - 1);
  const int ci = min(max((int)floorf(cp[comp_c] * scale), 0), W - 1);
  const long long cell = (long long)ri * W + ci;
  for (int ch = 0; ch < C; ++ch) {
    const float v = feat[(long long)b * fbstride + (long long)ch * KN + n];
    atomicMax((int*)&out[((long long)(b * C + ch)) * H * W + cell],
              __float_as_int(v));
  }
}

// ---------------------------------------------------------------------------
// 3x3 SAME conv as implicit-GEMM WMMA. Concat via (in0,C0)+(in1,C1).
// Block = (Cout/16) waves; each wave owns 16 out-ch x 16 pixels.
// Grid: (Wout/16, Hout, NB).
//
// Data path:
//   1) input slab (Cin x 3 x Wi, f32) -> LDS. For pad==0 (stride-2) convs the
//      slab is a rectangular 3-D tile, fetched with ONE Tensor-Data-Mover DMA
//      (tensor_load_to_lds + s_wait_tensorcnt); otherwise manual staging.
//   2) f16 im2col [pixel][K] built once in LDS (K contiguous per pixel), so
//      every WMMA B fragment is a pair of ds_load_b128.
//   3) K looped in chunks of 32 (Ktot = Cin*9 is always a multiple of 32 here)
//      with per-chunk weight staging.
// ---------------------------------------------------------------------------
__global__ void conv3x3_wmma_kernel(const float* __restrict__ in0, int C0,
                                    const float* __restrict__ in1, int C1,
                                    int H, int W, int stride, int pad,
                                    const float* __restrict__ wgt,
                                    const float* __restrict__ bias, int Cout,
                                    float* __restrict__ out, int Hout, int Wout,
                                    int do_relu) {
  __shared__ __align__(16) float    sInF[6336];      // max(64*3*33, 96*3*18)
  __shared__ __align__(16) _Float16 sCol[16 * 864];  // [pixel][K], K<=96*9
  __shared__ __align__(16) _Float16 sWk[64 * 32];    // per-chunk weights

  const int tid  = threadIdx.x;
  const int lane = tid & 31;
  const int wave = tid >> 5;
  const int nb  = blockIdx.z;
  const int oy  = blockIdx.y;
  const int ox0 = blockIdx.x * 16;
  const int Cin  = C0 + C1;
  const int Ktot = Cin * 9;
  const int Wi = 15 * stride + 3;

  bool staged = false;
#if TDM_OK
  if (pad == 0) {
    // Rectangular tile: X = Wi cols (stride 1), Y = 3 rows (stride W),
    // Z = Cin channels (stride H*W). Right/bottom overreach is zero-filled
    // via tensor_dim OOB handling.
    if (tid < 32) {  // one wave issues the DMA (TDM ignores EXEC)
      const int iy0 = oy * stride;
      const int ix0 = ox0 * stride;
      const unsigned long long ga = (unsigned long long)
          (const void*)(in0 + ((long long)(nb * C0) * H + iy0) * W + ix0);
      const unsigned ldsa = (unsigned)(unsigned long long)(const void*)&sInF[0];
      const unsigned td0 = (unsigned)(W - ix0);
      const unsigned td1 = (unsigned)(H - iy0);
      const unsigned long long s0 = (unsigned long long)W;        // dim0 stride
      const unsigned long long s1 = (unsigned long long)H * W;    // dim1 stride
      u32x4 g0;
      g0[0] = 1u;                                  // count=1, normal mode
      g0[1] = ldsa;                                // lds_addr
      g0[2] = (unsigned)(ga & 0xffffffffull);      // global_addr lo
      g0[3] = (unsigned)((ga >> 32) & 0x1ffffffull) | (2u << 30);  // type=2
      i32x8 g1;
      g1[0] = (int)(2u << 16);                     // data_size = 4B
      g1[1] = (int)((td0 & 0xffffu) << 16);        // tensor_dim0 lo16
      g1[2] = (int)(((td0 >> 16) & 0xffffu) | ((td1 & 0xffffu) << 16));
      g1[3] = (int)(((td1 >> 16) & 0xffffu) | ((unsigned)Wi << 16)); // tile_dim0
      g1[4] = (int)(3u | ((unsigned)Cin << 16));   // tile_dim1=3, tile_dim2=Cin
      g1[5] = (int)(s0 & 0xffffffffull);           // tensor_dim0_stride lo
      g1[6] = (int)(((s0 >> 32) & 0xffffull) | ((s1 & 0xffffull) << 16));
      g1[7] = (int)((s1 >> 16) & 0xffffffffull);   // tensor_dim1_stride hi
      i32x4 g2;
      g2[0] = Cin; g2[1] = 0; g2[2] = 0; g2[3] = 0; // tensor_dim2
      i32x4 g3;
      g3[0] = 0; g3[1] = 0; g3[2] = 0; g3[3] = 0;
      i32x8 g4;                                     // extra descriptor group
      g4[0] = 0; g4[1] = 0; g4[2] = 0; g4[3] = 0;   // (6-arg toolchain form)
      g4[4] = 0; g4[5] = 0; g4[6] = 0; g4[7] = 0;
      __builtin_amdgcn_tensor_load_to_lds(g0, g1, g2, g3, g4, 0);
      __builtin_amdgcn_s_wait_tensorcnt(0);
    }
    staged = true;
  }
#endif
  if (!staged) {
    for (int i = tid; i < Cin * 3 * Wi; i += blockDim.x) {
      const int ci  = i / (3 * Wi);
      const int r   = i - ci * 3 * Wi;
      const int ky  = r / Wi;
      const int ixl = r - ky * Wi;
      const int iy = oy * stride - pad + ky;
      const int ix = ox0 * stride - pad + ixl;
      float v = 0.0f;
      if (iy >= 0 && iy < H && ix >= 0 && ix < W) {
        v = (ci < C0)
              ? in0[((long long)(nb * C0 + ci) * H + iy) * W + ix]
              : in1[((long long)(nb * C1 + (ci - C0)) * H + iy) * W + ix];
      }
      sInF[(ci * 3 + ky) * Wi + ixl] = v;
    }
  }
  __syncthreads();

  // Build fragment-ready f16 im2col: sCol[pixel][K], K contiguous.
  for (int i = tid; i < 16 * Ktot; i += blockDim.x) {
    const int col = i / Ktot;
    const int k   = i - col * Ktot;
    const int ci = k / 9;
    const int rr = k - ci * 9;
    const int ky = rr / 3;
    const int kx = rr - ky * 3;
    sCol[col * Ktot + k] =
        (_Float16)sInF[(ci * 3 + ky) * Wi + col * stride + kx];
  }

  v8f acc = {};
  const int kbB = (lane & 16) ? 16 : 0;
  const _Float16* bcol = &sCol[(lane & 15) * Ktot];

  for (int k0 = 0; k0 < Ktot; k0 += 32) {
    __syncthreads();
    for (int i = tid; i < Cout * 32; i += blockDim.x)
      sWk[i] = (_Float16)wgt[(i >> 5) * Ktot + k0 + (i & 31)];
    __syncthreads();

    v16h a = load_a_frag(sWk, wave * 16, 0, 32, lane);
    v16h b = load_b_frag_kcontig(bcol + k0 + kbB);
    acc = __builtin_amdgcn_wmma_f32_16x16x32_f16(false, a, false, b,
                                                 (short)0, acc, false, false);
  }

#pragma unroll
  for (int r = 0; r < 8; ++r) {
    const int m = wave * 16 + r + ((lane & 16) ? 8 : 0);
    const int ox = ox0 + (lane & 15);
    if (ox < Wout && m < Cout) {
      float v = acc[r] + bias[m];
      if (do_relu) v = fmaxf(v, 0.0f);
      out[((long long)(nb * Cout + m) * Hout + oy) * Wout + ox] = v;
    }
  }
}

// ---------------------------------------------------------------------------
// movable_logit_2d = conv1x1(up2(shallow)); out (B,3,64,2048)
// ---------------------------------------------------------------------------
__global__ void rv_head_kernel(const float* __restrict__ shallow,
                               const float* __restrict__ w,
                               const float* __restrict__ bias,
                               float* __restrict__ out) {
  const int idx = blockIdx.x * blockDim.x + threadIdx.x;
  if (idx >= KB * 64 * 2048) return;
  const int x = idx & 2047;
  const int y = (idx >> 11) & 63;
  const int b = idx >> 17;
  const int sy = y >> 1, sx = x >> 1;
  float a0 = bias[0], a1 = bias[1], a2 = bias[2];
  for (int c = 0; c < 32; ++c) {
    const float s = shallow[((long long)(b * 32 + c) * 32 + sy) * 1024 + sx];
    a0 += w[c] * s;
    a1 += w[32 + c] * s;
    a2 += w[64 + c] * s;
  }
  out[((long long)(b * 3 + 0) * 64 + y) * 2048 + x] = a0;
  out[((long long)(b * 3 + 1) * 64 + y) * 2048 + x] = a1;
  out[((long long)(b * 3 + 2) * 64 + y) * 2048 + x] = a2;
}

// ---------------------------------------------------------------------------
// Bilinear gather: out (B,C,N) from fmap (B,C,H,W) at scaled coords (t=T-1).
// ---------------------------------------------------------------------------
__global__ void bilinear_kernel(const float* __restrict__ fmap, int C, int H,
                                int W, const float* __restrict__ coord,
                                long long cbstride, long long ctoff, int comp_r,
                                int comp_c, float scale, float* __restrict__ out) {
  const long long idx = (long long)blockIdx.x * blockDim.x + threadIdx.x;
  if (idx >= (long long)KB * KN) return;
  const int b = (int)(idx / KN);
  const int n = (int)(idx % KN);
  const float* cp = coord + (long long)b * cbstride + ctoff + (long long)n * 3;
  const float r = cp[comp_r] * scale;
  const float c = cp[comp_c] * scale;
  const float r0 = floorf(r), c0 = floorf(c);
  const float fr = r - r0, fc = c - c0;
  const int r0i = min(max((int)r0, 0), H - 1);
  const int r1i = min(r0i + 1, H - 1);
  const int c0i = min(max((int)c0, 0), W - 1);
  const int c1i = min(c0i + 1, W - 1);
  const float w00 = (1.f - fr) * (1.f - fc), w01 = (1.f - fr) * fc;
  const float w10 = fr * (1.f - fc), w11 = fr * fc;
  for (int ch = 0; ch < C; ++ch) {
    const float* fp = fmap + (long long)(b * C + ch) * H * W;
    out[(long long)(b * C + ch) * KN + n] =
        fp[r0i * W + c0i] * w00 + fp[r0i * W + c1i] * w01 +
        fp[r1i * W + c0i] * w10 + fp[r1i * W + c1i] * w11;
  }
}

// usum = x1b + up2(x2b)
__global__ void upsample_add_kernel(const float* __restrict__ a,
                                    const float* __restrict__ s,
                                    float* __restrict__ out, int C, int H,
                                    int W, long long total) {
  const long long i = (long long)blockIdx.x * blockDim.x + threadIdx.x;
  if (i >= total) return;
  const int x = (int)(i % W);
  const int y = (int)((i / W) % H);
  const long long nc = i / ((long long)W * H);
  out[i] = a[i] + s[(nc * (H / 2) + (y >> 1)) * (W / 2) + (x >> 1)];
}

// moving_logit_3d = fuse_w * concat(f_t0, mv3d) + fuse_b ; out (B,3,N)
__global__ void fuse_kernel(const float* __restrict__ f_t0, long long fbstride,
                            const float* __restrict__ mv3d,
                            const float* __restrict__ fw,
                            const float* __restrict__ fb,
                            float* __restrict__ out) {
  const long long idx = (long long)blockIdx.x * blockDim.x + threadIdx.x;
  if (idx >= (long long)KB * KN) return;
  const int b = (int)(idx / KN);
  const int n = (int)(idx % KN);
  float a0 = fb[0], a1 = fb[1], a2 = fb[2];
  for (int c = 0; c < 64; ++c) {
    const float v = f_t0[(long long)b * fbstride + (long long)c * KN + n];
    a0 += fw[c] * v;
    a1 += fw[96 + c] * v;
    a2 += fw[192 + c] * v;
  }
  for (int c = 0; c < 32; ++c) {
    const float v = mv3d[(long long)(b * 32 + c) * KN + n];
    a0 += fw[64 + c] * v;
    a1 += fw[96 + 64 + c] * v;
    a2 += fw[192 + 64 + c] * v;
  }
  out[((long long)(b * 3 + 0)) * KN + n] = a0;
  out[((long long)(b * 3 + 1)) * KN + n] = a1;
  out[((long long)(b * 3 + 2)) * KN + n] = a2;
}

// ---------------------------------------------------------------------------
extern "C" void kernel_launch(void* const* d_in, const int* in_sizes, int n_in,
                              void* d_out, int out_size, void* d_ws,
                              size_t ws_size, hipStream_t stream) {
  (void)in_sizes; (void)n_in; (void)out_size; (void)ws_size;
  const float* xyzi = (const float*)d_in[0];
  const float* des  = (const float*)d_in[1];
  const float* sph  = (const float*)d_in[2];
  // params sorted: bev_c1_b,bev_c1_w,bev_c2_b,bev_c2_w,bev_c3_b,bev_c3_w,
  // bev_c4_b,bev_c4_w,bev_c5_b,bev_c5_w,fuse_b,fuse_w,pn_b1,pn_b2,pn_w1,
  // pn_w2,rv_c1_b,rv_c1_w,rv_c2_b,rv_c2_w,rv_head_b,rv_head_w
  const float* bev_c1_b = (const float*)d_in[3];
  const float* bev_c1_w = (const float*)d_in[4];
  const float* bev_c2_b = (const float*)d_in[5];
  const float* bev_c2_w = (const float*)d_in[6];
  const float* bev_c3_b = (const float*)d_in[7];
  const float* bev_c3_w = (const float*)d_in[8];
  const float* bev_c4_b = (const float*)d_in[9];
  const float* bev_c4_w = (const float*)d_in[10];
  const float* bev_c5_b = (const float*)d_in[11];
  const float* bev_c5_w = (const float*)d_in[12];
  const float* fuse_b   = (const float*)d_in[13];
  const float* fuse_w   = (const float*)d_in[14];
  const float* pn_b1    = (const float*)d_in[15];
  const float* pn_b2    = (const float*)d_in[16];
  const float* pn_w1    = (const float*)d_in[17];
  const float* pn_w2    = (const float*)d_in[18];
  const float* rv_c1_b  = (const float*)d_in[19];
  const float* rv_c1_w  = (const float*)d_in[20];
  const float* rv_c2_b  = (const float*)d_in[21];
  const float* rv_c2_w  = (const float*)d_in[22];
  const float* rv_head_b = (const float*)d_in[23];
  const float* rv_head_w = (const float*)d_in[24];

  float* ws = (float*)d_ws;
  size_t o = 0;
  float* f       = ws + o; o += (size_t)KBT * 64 * KN;
  float* bev     = ws + o; o += (size_t)KB * 64 * 512 * 512;
  float* rv      = ws + o; o += (size_t)KB * 64 * 64 * 2048;
  float* shallow = ws + o; o += (size_t)KB * 32 * 32 * 1024;
  float* deep    = ws + o; o += (size_t)KB * 32 * 16 * 512;
  float* sh3d    = ws + o; o += (size_t)KB * 32 * KN;
  float* dp3d    = ws + o; o += (size_t)KB * 32 * KN;
  float* sh_bev  = ws + o; o += (size_t)KB * 32 * 256 * 256;
  float* dp_bev  = ws + o; o += (size_t)KB * 32 * 128 * 128;
  float* x1      = ws + o; o += (size_t)KB * 64 * 256 * 256;
  float* x1b     = ws + o; o += (size_t)KB * 64 * 256 * 256;
  float* x2      = ws + o; o += (size_t)KB * 64 * 128 * 128;
  float* x2b     = ws + o; o += (size_t)KB * 64 * 128 * 128;
  float* usum    = ws + o; o += (size_t)KB * 64 * 256 * 256;
  float* mfeat   = ws + o; o += (size_t)KB * 32 * 256 * 256;
  float* mv3d    = ws + o; o += (size_t)KB * 32 * KN;

  float* out_3d = (float*)d_out;                       // (B,3,N)
  float* out_2d = (float*)d_out + (size_t)KB * 3 * KN; // (B,3,64,2048)

  const long long cb = (long long)KT * KN * 3;       // coord batch stride
  const long long ct = (long long)(KT - 1) * KN * 3; // t = T-1 offset

  // Zero atomic-max targets (graph-capturable).
  (void)hipMemsetAsync(bev, 0, (size_t)KB * 64 * 512 * 512 * 4, stream);
  (void)hipMemsetAsync(rv, 0, (size_t)KB * 64 * 64 * 2048 * 4, stream);
  (void)hipMemsetAsync(sh_bev, 0, (size_t)KB * 32 * 256 * 256 * 4, stream);
  (void)hipMemsetAsync(dp_bev, 0, (size_t)KB * 32 * 128 * 128 * 4, stream);

  // PointNet (WMMA f16)
  pointnet_wmma_kernel<<<dim3(KN / 32, KBT), 256, 0, stream>>>(
      xyzi, pn_w1, pn_b1, pn_w2, pn_b2, f);

  // Pillarize
  scatter_bev_all_kernel<<<(KBT * KN + 255) / 256, 256, 0, stream>>>(f, des, bev);
  scatter_t0_kernel<<<(KB * KN + 255) / 256, 256, 0, stream>>>(
      f + (size_t)(KT - 1) * 64 * KN, (long long)KT * 64 * KN, 64,
      sph, cb, ct, 1, 0, 1.0f, 64, 2048, rv);

  // Range-view branch
  conv3x3_wmma_kernel<<<dim3(1024 / 16, 32, KB), 64, 0, stream>>>(
      rv, 64, nullptr, 0, 64, 2048, 2, 0, rv_c1_w, rv_c1_b, 32,
      shallow, 32, 1024, 1);
  conv3x3_wmma_kernel<<<dim3(512 / 16, 16, KB), 64, 0, stream>>>(
      shallow, 32, nullptr, 0, 32, 1024, 2, 0, rv_c2_w, rv_c2_b, 32,
      deep, 16, 512, 1);
  rv_head_kernel<<<(KB * 64 * 2048 + 255) / 256, 256, 0, stream>>>(
      shallow, rv_head_w, rv_head_b, out_2d);

  bilinear_kernel<<<(KB * KN + 255) / 256, 256, 0, stream>>>(
      shallow, 32, 32, 1024, sph, cb, ct, 1, 0, 0.5f, sh3d);
  bilinear_kernel<<<(KB * KN + 255) / 256, 256, 0, stream>>>(
      deep, 32, 16, 512, sph, cb, ct, 1, 0, 0.25f, dp3d);

  scatter_t0_kernel<<<(KB * KN + 255) / 256, 256, 0, stream>>>(
      sh3d, (long long)32 * KN, 32, des, cb, ct, 0, 1, 0.5f, 256, 256, sh_bev);
  scatter_t0_kernel<<<(KB * KN + 255) / 256, 256, 0, stream>>>(
      dp3d, (long long)32 * KN, 32, des, cb, ct, 0, 1, 0.25f, 128, 128, dp_bev);

  // BEV branch
  conv3x3_wmma_kernel<<<dim3(256 / 16, 256, KB), 128, 0, stream>>>(
      bev, 64, nullptr, 0, 512, 512, 2, 0, bev_c1_w, bev_c1_b, 64,
      x1, 256, 256, 1);
  conv3x3_wmma_kernel<<<dim3(256 / 16, 256, KB), 128, 0, stream>>>(
      x1, 64, sh_bev, 32, 256, 256, 1, 1, bev_c2_w, bev_c2_b, 64,
      x1b, 256, 256, 1);
  conv3x3_wmma_kernel<<<dim3(128 / 16, 128, KB), 128, 0, stream>>>(
      x1b, 64, nullptr, 0, 256, 256, 2, 0, bev_c3_w, bev_c3_b, 64,
      x2, 128, 128, 1);
  conv3x3_wmma_kernel<<<dim3(128 / 16, 128, KB), 128, 0, stream>>>(
      x2, 64, dp_bev, 32, 128, 128, 1, 1, bev_c4_w, bev_c4_b, 64,
      x2b, 128, 128, 1);

  const long long utotal = (long long)KB * 64 * 256 * 256;
  upsample_add_kernel<<<(int)((utotal + 255) / 256), 256, 0, stream>>>(
      x1b, x2b, usum, 64, 256, 256, utotal);
  conv3x3_wmma_kernel<<<dim3(256 / 16, 256, KB), 64, 0, stream>>>(
      usum, 64, nullptr, 0, 256, 256, 1, 1, bev_c5_w, bev_c5_b, 32,
      mfeat, 256, 256, 0);

  bilinear_kernel<<<(KB * KN + 255) / 256, 256, 0, stream>>>(
      mfeat, 32, 256, 256, des, cb, ct, 0, 1, 0.5f, mv3d);

  fuse_kernel<<<(KB * KN + 255) / 256, 256, 0, stream>>>(
      f + (size_t)(KT - 1) * 64 * KN, (long long)KT * 64 * KN, mv3d,
      fuse_w, fuse_b, out_3d);
}
